// NetworkGL_70033736728881
// MI455X (gfx1250) — matrix operand order repaired and checked by reading, not compiled, not run on hardware
//
#include <hip/hip_runtime.h>
#include <cstdint>
#include <cstddef>

// ---------------- model dims ----------------
#define Bm     32
#define Lm     2048
#define Cm     64
#define Pm     16
#define Nm     256
#define Hm     64
#define BCm    2048
#define PREDm  96
#define IHm    512
#define THIDm  256
#define DSLm   64

typedef __attribute__((ext_vector_type(16))) _Float16 v16h;
typedef __attribute__((ext_vector_type(8)))  float    v8f;

// ---- CDNA5 WMMA helpers (gfx1250, wave32) ----
__device__ __forceinline__ v8f wmma16(v16h a, v16h b, v8f c) {
  // v_wmma_f32_16x16x32_f16  D = A(16x32,f16) * B(32x16,f16) + C(f32)
  return __builtin_amdgcn_wmma_f32_16x16x32_f16(false, a, false, b, (short)0, c, false, false);
}
// A-matrix (16x32 f16) element k for lane/vector-element e (ISA 7.12.2 table)
__device__ __forceinline__ int a_kmap(int lane, int e) {
  int j = e >> 1;
  return ((j & 4) << 2) | ((lane >> 4) << 3) | ((j & 3) << 1) | (e & 1);
}
// B-matrix (32x16 f16, KxN) : same k order as A (ISA 7.12.2 table)
__device__ __forceinline__ int b_kmap(int lane, int e) {
  return a_kmap(lane, e);
}
__device__ __forceinline__ float gelu_f(float x) {
  return 0.5f * x * (1.0f + erff(x * 0.7071067811865475f));
}
__device__ __forceinline__ float sigmoid_f(float x) {
  return 1.0f / (1.0f + expf(-x));
}

// =====================================================================
// K1: RevIN stats + normalize + EMA decomposition (blocked scan) +
//     seasonal patching into h[BC,N,P] + downsampled trend
// grid (4,32): blockIdx.y=b, blockIdx.x=channel-group of 16
// 256 threads: tx%16 = channel-in-group, tx/16 = segment (128 l each)
// =====================================================================
__global__ __launch_bounds__(256) void k1_revin_ema(
    const float* __restrict__ x, const float* __restrict__ revin_w,
    const float* __restrict__ revin_b,
    float* __restrict__ h, float* __restrict__ tds,
    float* __restrict__ meanb, float* __restrict__ stdb) {
  const int tx = threadIdx.x;
  const int ci = tx & 15, seg = tx >> 4;
  const int b = blockIdx.y, cg = blockIdx.x;
  const int c = cg * 16 + ci, bc = b * Cm + c;
  __shared__ float redS[256], redQ[256], segend[256], carr[256];
  __shared__ float meanv[16], stdv[16];

  const float* xp = x + (size_t)b * Lm * Cm + c;
  float s = 0.f, ss = 0.f;
  for (int i = 0; i < 128; ++i) {
    float v = xp[(size_t)(seg * 128 + i) * Cm];
    s += v; ss += v * v;
  }
  redS[tx] = s; redQ[tx] = ss;
  __syncthreads();
  if (seg == 0) {
    float S = 0.f, Q = 0.f;
    for (int q = 0; q < 16; ++q) { S += redS[q * 16 + ci]; Q += redQ[q * 16 + ci]; }
    float mean = S / (float)Lm;
    float var = (Q - S * mean) / (float)(Lm - 1);   // ddof=1
    float st = sqrtf(fmaxf(var, 0.f)) + 1e-5f;
    meanv[ci] = mean; stdv[ci] = st;
    meanb[bc] = mean; stdb[bc] = st;
  }
  __syncthreads();
  const float mean = meanv[ci], st = stdv[ci];
  const float rw = revin_w[c], rb = revin_b[c];

  // pass 1: local recurrence with zero incoming carry
  float v = 0.f;
  for (int i = 0; i < 128; ++i) {
    int l = seg * 128 + i;
    float xn = (xp[(size_t)l * Cm] - mean) / st * rw + rb;
    v = (l == 0) ? xn : fmaf(0.8f, v, 0.2f * xn);
  }
  segend[tx] = v;
  __syncthreads();
  if (seg == 0) {
    const float dec = powf(0.8f, 128.0f);   // carry decay across one segment
    float cprev = 0.f;
    for (int sg = 0; sg < 16; ++sg) {
      carr[sg * 16 + ci] = cprev;
      cprev = segend[sg * 16 + ci] + dec * cprev;
    }
  }
  __syncthreads();

  // pass 2: exact trend with carry; emit seasonal patches + trend_ds
  v = carr[tx];
  float* hrow = h + (size_t)bc * (Nm * Pm);
  for (int i = 0; i < 128; ++i) {
    int l = seg * 128 + i;
    float xn = (xp[(size_t)l * Cm] - mean) / st * rw + rb;
    float tr = (l == 0) ? xn : fmaf(0.8f, v, 0.2f * xn);
    v = tr;
    float se = xn - tr;
    if ((l & 31) == 0) tds[(size_t)bc * DSLm + (l >> 5)] = tr;
    int n1 = l >> 3, p1 = l & 7;
    hrow[n1 * Pm + p1] = se;                          // p = 0..7 slot
    if (n1 > 0) hrow[(n1 - 1) * Pm + p1 + 8] = se;    // p = 8..15 slot
    if (l == Lm - 1)                                   // replication padding
      for (int pp = 8; pp < 16; ++pp) hrow[(Nm - 1) * Pm + pp] = se;
  }
}

// =====================================================================
// K2a: inter-patch branch: pooled -> MLP(256->512->256) -> sigmoid
// grid 128 blocks of 16 bc-rows, 256 threads (8 waves)
// =====================================================================
__global__ __launch_bounds__(256) void k2a_global(
    const float* __restrict__ hbuf, float* __restrict__ wgt,
    const float* __restrict__ mlp1_w, const float* __restrict__ mlp1_b,
    const float* __restrict__ mlp2_w, const float* __restrict__ mlp2_b) {
  const int tx = threadIdx.x, w = tx >> 5, lane = tx & 31, col = lane & 15;
  const int bc0 = blockIdx.x * 16;
  __shared__ _Float16 pooled[16 * Nm];     // A for GEMM1
  __shared__ _Float16 act1[16 * IHm];      // A for GEMM2

  for (int idx = tx; idx < 16 * Nm; idx += 256) {
    int i = idx >> 8, n = idx & 255;
    const float4* p4 = (const float4*)(hbuf + (size_t)(bc0 + i) * (Nm * Pm) + n * Pm);
    float4 a0 = p4[0], a1 = p4[1], a2 = p4[2], a3 = p4[3];
    float sum = a0.x + a0.y + a0.z + a0.w + a1.x + a1.y + a1.z + a1.w +
                a2.x + a2.y + a2.z + a2.w + a3.x + a3.y + a3.z + a3.w;
    pooled[i * Nm + n] = (_Float16)(sum * (1.f / 16.f));
  }
  __syncthreads();

  // GEMM1: [16,256] x [256,512]
  for (int t = 0; t < 4; ++t) {
    int nt = w * 4 + t;
    v8f acc = {};
    for (int ks = 0; ks < 8; ++ks) {
      v16h a, bb;
#pragma unroll
      for (int e = 0; e < 16; ++e) {
        a[e]  = pooled[col * Nm + ks * 32 + a_kmap(lane, e)];
        bb[e] = (_Float16)mlp1_w[(size_t)(nt * 16 + col) * Nm + ks * 32 + b_kmap(lane, e)];
      }
      acc = wmma16(a, bb, acc);
    }
#pragma unroll
    for (int r = 0; r < 8; ++r) {
      int m = r + ((lane >> 4) << 3);
      int j = nt * 16 + col;
      act1[m * IHm + j] = (_Float16)gelu_f(acc[r] + mlp1_b[j]);
    }
  }
  __syncthreads();

  // GEMM2: [16,512] x [512,256] -> sigmoid -> wgt
  for (int t = 0; t < 2; ++t) {
    int nt = w * 2 + t;
    v8f acc = {};
    for (int ks = 0; ks < 16; ++ks) {
      v16h a, bb;
#pragma unroll
      for (int e = 0; e < 16; ++e) {
        a[e]  = act1[col * IHm + ks * 32 + a_kmap(lane, e)];
        bb[e] = (_Float16)mlp2_w[(size_t)(nt * 16 + col) * IHm + ks * 32 + b_kmap(lane, e)];
      }
      acc = wmma16(a, bb, acc);
    }
#pragma unroll
    for (int r = 0; r < 8; ++r) {
      int m = r + ((lane >> 4) << 3);
      int j = nt * 16 + col;
      wgt[(size_t)(bc0 + m) * Nm + j] = sigmoid_f(acc[r] + mlp2_b[j]);
    }
  }
}

// =====================================================================
// K2b: fused local branch + gated fusion + LayerNorm (in-place on h)
// grid (4,2048): blockIdx.y=bc, blockIdx.x=chunk of 64 patch rows
// =====================================================================
__global__ __launch_bounds__(256) void k2b_local(
    float* __restrict__ hbuf, const float* __restrict__ wgt,
    const float* __restrict__ fc1_w, const float* __restrict__ fc1_b,
    const float* __restrict__ bn1_w, const float* __restrict__ bn1_b,
    const float* __restrict__ bn1_rm, const float* __restrict__ bn1_rv,
    const float* __restrict__ conv_w, const float* __restrict__ conv_b,
    const float* __restrict__ bn2_w, const float* __restrict__ bn2_b,
    const float* __restrict__ bn2_rm, const float* __restrict__ bn2_rv,
    const float* __restrict__ fc2_w, const float* __restrict__ fc2_b,
    const float* __restrict__ scale_p,
    const float* __restrict__ gate_w, const float* __restrict__ gate_b,
    const float* __restrict__ ln_w, const float* __restrict__ ln_b) {
  const int tx = threadIdx.x, w = tx >> 5, lane = tx & 31, col = lane & 15;
  const int bc = blockIdx.y;
  const int rbase = blockIdx.x * 64;

  __shared__ float   hs[64 * 16];
  __shared__ float   us[64 * 64];       // fc1 output (f32, pre-conv); reused as z
  __shared__ _Float16 uh[64 * 64];      // conv output as WMMA-A f16
  __shared__ float   localb[64 * 16];
  __shared__ float   wf1[64 * 16], wf2[16 * 64], wg[16 * 32];
  __shared__ float   b1s[64], b2s[16], bgs[16], lnws[16], lnbs[16];
  __shared__ float   s1[64], o1[64], s2[64], o2[64];
  __shared__ float   cw0[64], cw1[64], cw2[64], cbs[64], gfac[64];
  float* zb = us;   // us is dead after the conv stage

  float* hrow = hbuf + (size_t)bc * (Nm * Pm) + rbase * Pm;

  for (int i = tx; i < 1024; i += 256) { hs[i] = hrow[i]; wf1[i] = fc1_w[i]; wf2[i] = fc2_w[i]; }
  for (int i = tx; i < 512; i += 256) wg[i] = gate_w[i];
  if (tx < 64) {
    b1s[tx] = fc1_b[tx];
    int gn = rbase + tx;
    float si1 = bn1_w[gn] * rsqrtf(bn1_rv[gn] + 1e-5f);
    s1[tx] = si1; o1[tx] = bn1_b[gn] - bn1_rm[gn] * si1;
    float si2 = bn2_w[gn] * rsqrtf(bn2_rv[gn] + 1e-5f);
    s2[tx] = si2; o2[tx] = bn2_b[gn] - bn2_rm[gn] * si2;
    cw0[tx] = conv_w[gn * 3 + 0]; cw1[tx] = conv_w[gn * 3 + 1]; cw2[tx] = conv_w[gn * 3 + 2];
    cbs[tx] = conv_b[gn];
    gfac[tx] = 1.0f + scale_p[0] * wgt[(size_t)bc * Nm + gn];
  }
  if (tx < 16) { b2s[tx] = fc2_b[tx]; bgs[tx] = gate_b[tx]; lnws[tx] = ln_w[tx]; lnbs[tx] = ln_b[tx]; }
  __syncthreads();

  // fc1: [64,16(pad32)] x [16,64] -> gelu -> bn1 -> us ; 16 tiles, 2/wave
  for (int t = 0; t < 2; ++t) {
    int tile = w * 2 + t, mt = tile >> 2, nt = tile & 3;
    v16h a, bb;
#pragma unroll
    for (int e = 0; e < 16; ++e) {
      int ka = a_kmap(lane, e);
      a[e] = (ka < 16) ? (_Float16)hs[(mt * 16 + col) * 16 + ka] : (_Float16)0.f;
      int kb = b_kmap(lane, e);
      bb[e] = (kb < 16) ? (_Float16)wf1[(nt * 16 + col) * 16 + kb] : (_Float16)0.f;
    }
    v8f acc = {};
    acc = wmma16(a, bb, acc);
#pragma unroll
    for (int r = 0; r < 8; ++r) {
      int m = mt * 16 + r + ((lane >> 4) << 3);
      int j = nt * 16 + col;
      float v = gelu_f(acc[r] + b1s[j]);
      us[m * 64 + j] = v * s1[m] + o1[m];
    }
  }
  __syncthreads();

  // depthwise conv(3) along H (zero pad) -> gelu -> bn2 -> uh (f16)
  {
    int r = tx >> 2, jq = (tx & 3) * 16;
    float c0 = cw0[r], c1 = cw1[r], c2 = cw2[r], cb = cbs[r];
#pragma unroll
    for (int q = 0; q < 16; ++q) {
      int j = jq + q;
      float um1 = (j > 0) ? us[r * 64 + j - 1] : 0.f;
      float u0 = us[r * 64 + j];
      float up1 = (j < 63) ? us[r * 64 + j + 1] : 0.f;
      float v = gelu_f(c0 * um1 + c1 * u0 + c2 * up1 + cb);
      uh[r * 64 + j] = (_Float16)(v * s2[r] + o2[r]);
    }
  }
  __syncthreads();

  // fc2: [64,64] x [64,16] + bias + residual -> localb ; waves 0..3
  if (w < 4) {
    int mt = w;
    v8f acc = {};
    for (int ks = 0; ks < 2; ++ks) {
      v16h a, bb;
#pragma unroll
      for (int e = 0; e < 16; ++e) {
        a[e]  = uh[(mt * 16 + col) * 64 + ks * 32 + a_kmap(lane, e)];
        bb[e] = (_Float16)wf2[col * 64 + ks * 32 + b_kmap(lane, e)];
      }
      acc = wmma16(a, bb, acc);
    }
#pragma unroll
    for (int r = 0; r < 8; ++r) {
      int m = mt * 16 + r + ((lane >> 4) << 3);
      localb[m * 16 + col] = acc[r] + b2s[col] + hs[m * 16 + col];
    }
  }
  __syncthreads();

  // gate: A = [local | glob] [64,32] x gate_w^T [32,16] -> sigmoid -> fuse -> zb
  if (w < 4) {
    int mt = w;
    v16h a, bb;
#pragma unroll
    for (int e = 0; e < 16; ++e) {
      int ka = a_kmap(lane, e);
      int m = mt * 16 + col;
      a[e] = (_Float16)((ka < 16) ? localb[m * 16 + ka]
                                  : hs[m * 16 + (ka - 16)] * gfac[m]);
      bb[e] = (_Float16)wg[col * 32 + b_kmap(lane, e)];
    }
    v8f acc = {};
    acc = wmma16(a, bb, acc);
#pragma unroll
    for (int r = 0; r < 8; ++r) {
      int m = mt * 16 + r + ((lane >> 4) << 3);
      float g = sigmoid_f(acc[r] + bgs[col]);
      float hv = hs[m * 16 + col];
      zb[m * 16 + col] = hv + g * localb[m * 16 + col] + (1.f - g) * hv * gfac[m];
    }
  }
  __syncthreads();

  // LayerNorm over P=16, write back h
  if (tx < 64) {
    int m = tx;
    float mu = 0.f;
#pragma unroll
    for (int p = 0; p < 16; ++p) mu += zb[m * 16 + p];
    mu *= (1.f / 16.f);
    float var = 0.f;
#pragma unroll
    for (int p = 0; p < 16; ++p) { float d = zb[m * 16 + p] - mu; var += d * d; }
    var *= (1.f / 16.f);
    float inv = rsqrtf(var + 1e-5f);
#pragma unroll
    for (int p = 0; p < 16; ++p)
      hrow[m * 16 + p] = (zb[m * 16 + p] - mu) * inv * lnws[p] + lnbs[p];
  }
}

// =====================================================================
// K3: seasonal head GEMM (K=4096) with double-buffered ASYNC global->LDS
//     A-staging (ASYNCcnt pipeline) + trend MLP head + RevIN denorm
// grid 128 blocks of 16 bc-rows, 256 threads
// =====================================================================
#define K3_CH   256                   // K-chunk (floats per row per chunk)
#define K3_NCH  (Nm * Pm / K3_CH)     // 16 chunks
__global__ __launch_bounds__(256) void k3_heads(
    const float* __restrict__ hbuf, const float* __restrict__ tds,
    const float* __restrict__ meanb, const float* __restrict__ stdb,
    const float* __restrict__ revin_w, const float* __restrict__ revin_b,
    const float* __restrict__ seas_w, const float* __restrict__ seas_b,
    const float* __restrict__ t1w, const float* __restrict__ t1b,
    const float* __restrict__ t2w, const float* __restrict__ t2b,
    float* __restrict__ out) {
  const int tx = threadIdx.x, w = tx >> 5, lane = tx & 31, col = lane & 15;
  const int bc0 = blockIdx.x * 16;
  __shared__ float    AbufF[2][16 * K3_CH];   // double-buffered async A staging
  __shared__ _Float16 Tb[16 * DSLm];
  __shared__ _Float16 actT[16 * THIDm];
  __shared__ float    S[16 * PREDm];

  for (int idx = tx; idx < 16 * DSLm; idx += 256)
    Tb[idx] = (_Float16)tds[(size_t)(bc0 + (idx >> 6)) * DSLm + (idx & 63)];

  const unsigned long long hbase = (unsigned long long)(uintptr_t)hbuf;

  // issue one chunk's async copies: 16 rows x 256 f32 = 1024 x 16B, 4 per thread
  auto issue_chunk = [&](int c, int buf) {
#pragma unroll
    for (int q = 0; q < 4; ++q) {
      int idx16 = tx + q * 256;                 // 0..1023
      int i = idx16 >> 6;                       // row 0..15
      int kk = (idx16 & 63) << 2;               // float offset in row chunk
      unsigned int ldsoff = (unsigned int)(uintptr_t)(&AbufF[buf][i * K3_CH + kk]);
      unsigned int voff =
          (unsigned int)((((bc0 + i) * (Nm * Pm)) + c * K3_CH + kk) * 4);
      asm volatile("global_load_async_to_lds_b128 %0, %1, %2"
                   :: "v"(ldsoff), "v"(voff), "s"(hbase) : "memory");
    }
  };

  // seasonal: [16,4096] x [4096,96]; async double-buffered K pipeline
  issue_chunk(0, 0);
  v8f acc = {};
  for (int c = 0; c < K3_NCH; ++c) {
    if (c + 1 < K3_NCH) {
      issue_chunk(c + 1, (c + 1) & 1);
      asm volatile("s_wait_asynccnt 0x4" ::: "memory");  // chunk c landed (in-order)
    } else {
      asm volatile("s_wait_asynccnt 0x0" ::: "memory");
    }
    __syncthreads();   // all waves' portions of chunk c visible
    if (w < 6) {
      if (c + 1 < K3_NCH)   // prefetch next chunk of the B (seas_w) stream into GL2
        __builtin_prefetch(seas_w + (size_t)(w * 16 + col) * (Nm * Pm) +
                               (c + 1) * K3_CH + ((lane >> 4) << 7), 0, 3);
      const float* Ab = AbufF[c & 1];
      for (int ks = 0; ks < K3_CH / 32; ++ks) {
        v16h a, bb;
#pragma unroll
        for (int e = 0; e < 16; ++e) {
          a[e]  = (_Float16)Ab[col * K3_CH + ks * 32 + a_kmap(lane, e)];
          bb[e] = (_Float16)seas_w[(size_t)(w * 16 + col) * (Nm * Pm) + c * K3_CH + ks * 32 + b_kmap(lane, e)];
        }
        acc = wmma16(a, bb, acc);
      }
    }
    __syncthreads();   // buffer (c&1) free for reuse next-next iteration
  }
  if (w < 6) {
#pragma unroll
    for (int r = 0; r < 8; ++r) {
      int m = r + ((lane >> 4) << 3);
      int j = w * 16 + col;
      S[m * PREDm + j] = acc[r] + seas_b[j];
    }
  }

  // trend head GEMM1: [16,64] x [64,256] -> gelu
  for (int t = 0; t < 2; ++t) {
    int nt = w * 2 + t;
    v8f a2 = {};
    for (int ks = 0; ks < 2; ++ks) {
      v16h a, bb;
#pragma unroll
      for (int e = 0; e < 16; ++e) {
        a[e]  = Tb[col * DSLm + ks * 32 + a_kmap(lane, e)];
        bb[e] = (_Float16)t1w[(size_t)(nt * 16 + col) * DSLm + ks * 32 + b_kmap(lane, e)];
      }
      a2 = wmma16(a, bb, a2);
    }
#pragma unroll
    for (int r = 0; r < 8; ++r) {
      int m = r + ((lane >> 4) << 3);
      int j = nt * 16 + col;
      actT[m * THIDm + j] = (_Float16)gelu_f(a2[r] + t1b[j]);
    }
  }
  __syncthreads();

  // trend head GEMM2: [16,256] x [256,96], accumulate into S
  if (w < 6) {
    v8f a3 = {};
    for (int ks = 0; ks < 8; ++ks) {
      v16h a, bb;
#pragma unroll
      for (int e = 0; e < 16; ++e) {
        a[e]  = actT[col * THIDm + ks * 32 + a_kmap(lane, e)];
        bb[e] = (_Float16)t2w[(size_t)(w * 16 + col) * THIDm + ks * 32 + b_kmap(lane, e)];
      }
      a3 = wmma16(a, bb, a3);
    }
#pragma unroll
    for (int r = 0; r < 8; ++r) {
      int m = r + ((lane >> 4) << 3);
      int j = w * 16 + col;
      S[m * PREDm + j] += a3[r] + t2b[j];
    }
  }
  __syncthreads();

  // RevIN denorm + transpose store: out[b, pred, c]
  for (int idx = tx; idx < 16 * PREDm; idx += 256) {
    int i = idx / PREDm, pr = idx % PREDm;
    int bc = bc0 + i, bb2 = bc >> 6, cc = bc & 63;
    float v = S[i * PREDm + pr];
    out[((size_t)bb2 * PREDm + pr) * Cm + cc] =
        (v - revin_b[cc]) / revin_w[cc] * stdb[bc] + meanb[bc];
  }
}

// =====================================================================
extern "C" void kernel_launch(void* const* d_in, const int* in_sizes, int n_in,
                              void* d_out, int out_size, void* d_ws, size_t ws_size,
                              hipStream_t stream) {
  (void)in_sizes; (void)n_in; (void)out_size; (void)ws_size;
  const float* x        = (const float*)d_in[0];
  const float* revin_w  = (const float*)d_in[1];
  const float* revin_b  = (const float*)d_in[2];
  const float* fc1_w    = (const float*)d_in[3];
  const float* fc1_b    = (const float*)d_in[4];
  const float* bn1_w    = (const float*)d_in[5];
  const float* bn1_b    = (const float*)d_in[6];
  const float* bn1_rm   = (const float*)d_in[7];
  const float* bn1_rv   = (const float*)d_in[8];
  const float* conv_w   = (const float*)d_in[9];
  const float* conv_b   = (const float*)d_in[10];
  const float* bn2_w    = (const float*)d_in[11];
  const float* bn2_b    = (const float*)d_in[12];
  const float* bn2_rm   = (const float*)d_in[13];
  const float* bn2_rv   = (const float*)d_in[14];
  const float* fc2_w    = (const float*)d_in[15];
  const float* fc2_b    = (const float*)d_in[16];
  const float* mlp1_w   = (const float*)d_in[17];
  const float* mlp1_b   = (const float*)d_in[18];
  const float* mlp2_w   = (const float*)d_in[19];
  const float* mlp2_b   = (const float*)d_in[20];
  const float* scale    = (const float*)d_in[21];
  const float* gate_w   = (const float*)d_in[22];
  const float* gate_b   = (const float*)d_in[23];
  const float* ln_w     = (const float*)d_in[24];
  const float* ln_b     = (const float*)d_in[25];
  const float* seas_w   = (const float*)d_in[26];
  const float* seas_b   = (const float*)d_in[27];
  const float* trend1_w = (const float*)d_in[28];
  const float* trend1_b = (const float*)d_in[29];
  const float* trend2_w = (const float*)d_in[30];
  const float* trend2_b = (const float*)d_in[31];

  // workspace layout (floats): h | wgt | trend_ds | mean | std  (~36.2 MB)
  float* ws    = (float*)d_ws;
  float* h     = ws;                                 // BC*N*P = 8,388,608
  float* wgt   = h + (size_t)BCm * Nm * Pm;          // BC*N   =   524,288
  float* tds   = wgt + (size_t)BCm * Nm;             // BC*64  =   131,072
  float* meanb = tds + (size_t)BCm * DSLm;           // BC
  float* stdb  = meanb + BCm;                        // BC

  k1_revin_ema<<<dim3(4, Bm), 256, 0, stream>>>(x, revin_w, revin_b, h, tds, meanb, stdb);

  for (int k = 0; k < 2; ++k) {
    k2a_global<<<BCm / 16, 256, 0, stream>>>(
        h, wgt, mlp1_w + (size_t)k * IHm * Nm, mlp1_b + (size_t)k * IHm,
        mlp2_w + (size_t)k * Nm * IHm, mlp2_b + (size_t)k * Nm);
    k2b_local<<<dim3(4, BCm), 256, 0, stream>>>(
        h, wgt,
        fc1_w + (size_t)k * Hm * Pm, fc1_b + (size_t)k * Hm,
        bn1_w + (size_t)k * Nm, bn1_b + (size_t)k * Nm,
        bn1_rm + (size_t)k * Nm, bn1_rv + (size_t)k * Nm,
        conv_w + (size_t)k * Nm * 3, conv_b + (size_t)k * Nm,
        bn2_w + (size_t)k * Nm, bn2_b + (size_t)k * Nm,
        bn2_rm + (size_t)k * Nm, bn2_rv + (size_t)k * Nm,
        fc2_w + (size_t)k * Pm * Hm, fc2_b + (size_t)k * Pm,
        scale + k,
        gate_w + (size_t)k * Pm * 2 * Pm, gate_b + (size_t)k * Pm,
        ln_w + (size_t)k * Pm, ln_b + (size_t)k * Pm);
  }

  k3_heads<<<BCm / 16, 256, 0, stream>>>(
      h, tds, meanb, stdb, revin_w, revin_b,
      seas_w, seas_b, trend1_w, trend1_b, trend2_w, trend2_b, (float*)d_out);
}
